// GraphAttnBiasSpatial_55327768708486
// MI455X (gfx1250) — compile-verified, matrix operand order checked by reading, not run on hardware
//
#include <hip/hip_runtime.h>

typedef __attribute__((ext_vector_type(16))) _Float16 v16h;
typedef __attribute__((ext_vector_type(8)))  float    v8f;

#define N_NODE   50
#define NPAIR    2500     // 50*50
#define HID      64
#define HEADS    8
#define NP1      51
#define F_FRAMES 64
#define N_GRAPH  2048     // B*F = 32*64
#define NTILE    157      // ceil(2500/16)
#define CORE_STRIDE 2512  // 16 * NTILE, padded so tail tile stores unguarded

// ---------------------------------------------------------------------------
// Kernel 1: precompute hbc[h*2500 + p] = edge_emb[clip(dm[p])][h] + b2[h]
// (shared by all 2048 graphs; 80 KB, stays hot in L2)
// ---------------------------------------------------------------------------
__global__ void hbc_kernel(const int* __restrict__ dm,
                           const float* __restrict__ edge_emb,
                           const float* __restrict__ b2,
                           float* __restrict__ hbc) {
  int t = blockIdx.x * blockDim.x + threadIdx.x;
  if (t >= HEADS * NPAIR) return;
  int h = t / NPAIR, p = t % NPAIR;
  int d = dm[p];
  d = d < -1 ? -1 : (d > 4 ? 4 : d);   // clip(-1, MAX_HOP)
  if (d == -1) d = 0;                  // where(dm==-1, 0, dm)
  hbc[t] = edge_emb[d * HEADS + h] + b2[h];
}

// ---------------------------------------------------------------------------
// Kernel 2: borders. out[n,h,i,0] = out[n,h,0,i] = virt[h,i-1]; out[n,h,0,0]=0
// ---------------------------------------------------------------------------
__global__ void border_kernel(const float* __restrict__ virt,
                              float* __restrict__ out) {
  int t = blockIdx.x * blockDim.x + threadIdx.x;
  if (t >= N_GRAPH * HEADS * NP1) return;
  int i  = t % NP1;
  int nh = t / NP1;
  int h  = nh % HEADS;
  int n  = nh / HEADS;
  float v = (i == 0) ? 0.0f : virt[h * N_NODE + (i - 1)];
  size_t base = ((size_t)n * HEADS + h) * (size_t)(NP1 * NP1);
  out[base + (size_t)i * NP1] = v;   // column 0
  out[base + i]               = v;   // row 0
}

// ---------------------------------------------------------------------------
// Kernel 3: per-graph dist -> normalize -> MLP head projection via WMMA,
// staged through LDS for a fully coalesced store sweep.
// One block (8 waves, 256 threads) per graph.
// ---------------------------------------------------------------------------
__global__ void __launch_bounds__(256)
graph_attn_kernel(const float* __restrict__ x,
                  const float* __restrict__ w1,
                  const float* __restrict__ b1,
                  const float* __restrict__ w2,
                  const float* __restrict__ hbc,
                  float* __restrict__ out) {
  __shared__ __align__(16) float s_core[HEADS * CORE_STRIDE]; // 80384 B
  __shared__ float s_dist[NPAIR];          // 10000 B
  __shared__ float s_xg[3 * N_NODE];       // [c][node]
  __shared__ float s_w1[HID], s_b1[HID];
  __shared__ float s_w2[HEADS * HID];
  __shared__ float s_red[256], s_red2[256];
  __shared__ float s_stats[2];

  const int n   = blockIdx.x;
  const int b   = n / F_FRAMES;
  const int f   = n % F_FRAMES;
  const int tid = threadIdx.x;

  // stage joint coords: xg[node,c] = x[b, c, f, node%25, node/25]
  if (tid < 3 * N_NODE) {
    int c = tid / N_NODE, node = tid % N_NODE;
    int w = node / 25, hj = node % 25;
    s_xg[c * N_NODE + node] =
        x[(size_t)b * 9600 + (size_t)c * 3200 + f * 50 + hj * 2 + w];
  }
  if (tid < HID) { s_w1[tid] = w1[tid]; s_b1[tid] = b1[tid]; }
  for (int k = tid; k < HEADS * HID; k += 256) s_w2[k] = w2[k];
  __syncthreads();

  // pairwise distances + running moments
  float lsum = 0.0f, lsq = 0.0f;
  for (int p = tid; p < NPAIR; p += 256) {
    int i = p / N_NODE, j = p % N_NODE;
    float dx = s_xg[i]             - s_xg[j];
    float dy = s_xg[N_NODE + i]    - s_xg[N_NODE + j];
    float dz = s_xg[2*N_NODE + i]  - s_xg[2*N_NODE + j];
    float ss = dx*dx + dy*dy + dz*dz;
    float d  = ss > 0.0f ? sqrtf(ss) : 0.0f;
    s_dist[p] = d;
    lsum += d; lsq += d * d;
  }
  s_red[tid] = lsum; s_red2[tid] = lsq;
  __syncthreads();
  for (int s = 128; s > 0; s >>= 1) {
    if (tid < s) { s_red[tid] += s_red[tid + s]; s_red2[tid] += s_red2[tid + s]; }
    __syncthreads();
  }
  if (tid == 0) {
    float sum = s_red[0], sq = s_red2[0];
    float mean = sum * (1.0f / NPAIR);
    float var  = (sq - sum * sum * (1.0f / NPAIR)) * (1.0f / (NPAIR - 1)); // ddof=1
    var = var > 0.0f ? var : 0.0f;
    s_stats[0] = mean;
    s_stats[1] = 1.0f / (sqrtf(var) + 1e-5f);
  }
  __syncthreads();
  const float mean = s_stats[0];
  const float inv  = s_stats[1];

  const int wave = tid >> 5;
  const int lane = tid & 31;
  const int m16  = lane & 15;            // A-row / D-col within tile
  const int kb   = (lane >> 4) << 3;     // K-block select per lane half (0 or 8)

  // B matrix = w2^T padded 8->16 cols, constant across tiles.
  // Layout (32x16 f16, K x N): lane holds col N=lane%16,
  // element e -> K = (e%8) + 16*(e/8) + 8*(lane/16).
  v16h bm0, bm1;
#pragma unroll
  for (int e = 0; e < 16; ++e) {
    int k = (e & 7) + ((e >> 3) << 4) + kb;
    float v0 = (m16 < HEADS) ? s_w2[m16 * HID + k]      : 0.0f;
    float v1 = (m16 < HEADS) ? s_w2[m16 * HID + k + 32] : 0.0f;
    bm0[e] = (_Float16)v0;
    bm1[e] = (_Float16)v1;
  }

  // Sweep 16-row tiles of the 2500-pair index; wave-uniform loop (EXEC all 1s).
  for (int tile = wave; tile < NTILE; tile += 8) {
    int p = tile * 16 + m16;
    float dn = (p < NPAIR) ? (s_dist[p] - mean) * inv : 0.0f;

    // A tiles (16x32 f16): hcol[k] = relu(dn*w1[k] + b1[k]) for this lane's K set
    v16h a0, a1;
#pragma unroll
    for (int e = 0; e < 16; ++e) {
      int k = (e & 7) + ((e >> 3) << 4) + kb;
      float h0 = dn * s_w1[k]      + s_b1[k];
      float h1 = dn * s_w1[k + 32] + s_b1[k + 32];
      h0 = h0 > 0.0f ? h0 : 0.0f;
      h1 = h1 > 0.0f ? h1 : 0.0f;
      a0[e] = (_Float16)h0;
      a1[e] = (_Float16)h1;
    }

    v8f c = {};
    c = __builtin_amdgcn_wmma_f32_16x16x32_f16(false, a0, false, bm0,
                                               (short)0, c, false, false);
    c = __builtin_amdgcn_wmma_f32_16x16x32_f16(false, a1, false, bm1,
                                               (short)0, c, false, false);

    // D (16x16 f32): lane col N=m16 (=head), VGPR r -> row M = r + 8*(lane/16).
    // Rows r=0..7 are consecutive pair indices -> two b128 LDS stores, no guards
    // (CORE_STRIDE padding absorbs the tail tile).
    if (m16 < HEADS) {
      float4* dst = (float4*)&s_core[m16 * CORE_STRIDE + tile * 16 + kb];
      dst[0] = make_float4(c[0], c[1], c[2], c[3]);
      dst[1] = make_float4(c[4], c[5], c[6], c[7]);
    }
  }
  __syncthreads();

  // Coalesced epilogue: out[n,h,i+1,j+1] = core + hbc.
  // Flat offset h*2601 + (i+1)*51 + (j+1) = h*2601 + p + p/50 + 52.
  const size_t outbase = (size_t)n * (HEADS * NP1 * NP1);
  for (int t = tid; t < HEADS * NPAIR; t += 256) {
    int h = t / NPAIR;
    int p = t - h * NPAIR;
    int i = p / N_NODE;
    out[outbase + h * (NP1 * NP1) + p + i + (NP1 + 1)] =
        s_core[h * CORE_STRIDE + p] + hbc[t];
  }
}

// ---------------------------------------------------------------------------
extern "C" void kernel_launch(void* const* d_in, const int* in_sizes, int n_in,
                              void* d_out, int out_size, void* d_ws, size_t ws_size,
                              hipStream_t stream) {
  const float* x        = (const float*)d_in[0];
  const float* w1       = (const float*)d_in[1];
  const float* b1       = (const float*)d_in[2];
  const float* w2       = (const float*)d_in[3];
  const float* b2       = (const float*)d_in[4];
  const float* edge_emb = (const float*)d_in[5];
  const float* virt     = (const float*)d_in[6];
  const int*   dm       = (const int*)d_in[7];
  float* out = (float*)d_out;
  float* hbc = (float*)d_ws;   // 8*2500 floats = 80 KB scratch

  hbc_kernel<<<(HEADS * NPAIR + 255) / 256, 256, 0, stream>>>(dm, edge_emb, b2, hbc);
  border_kernel<<<(N_GRAPH * HEADS * NP1 + 255) / 256, 256, 0, stream>>>(virt, out);
  graph_attn_kernel<<<N_GRAPH, 256, 0, stream>>>(x, w1, b1, w2, hbc, out);
}
